// BCGNN_15934328668763
// MI455X (gfx1250) — matrix operand order repaired and checked by reading, not compiled, not run on hardware
//
#include <hip/hip_runtime.h>
#include <hip/hip_bf16.h>
#include <cstdint>
#include <cstddef>

#define NN 50000
#define EE 800000
#define HD 128
#define HD2 256
#define HD3 384
#define NITER 2
#define HSTRIDE 264   // LDS row stride in halves (16B aligned, conflict-friendly)

typedef __attribute__((ext_vector_type(16))) _Float16 v16h;
typedef __attribute__((ext_vector_type(8)))  float    v8f;

__device__ __forceinline__ v8f wmma16(v16h a, v16h b, v8f c) {
  // v_wmma_f32_16x16x32_f16: D = A(16x32 f16) * B(32x16 f16) + C(16x16 f32)
  return __builtin_amdgcn_wmma_f32_16x16x32_f16(false, a, false, b, (short)0, c, false, false);
}

// A-fragment (16x32 f16) from a row-major f32 row: elems 0..15 <- K=16*hi+i
__device__ __forceinline__ v16h a_frag_f32(const float* p, int hi) {
  v16h a;
  const float* p0 = p + 16 * hi;
  const float* p1 = p0 + 8;
#pragma unroll
  for (int i = 0; i < 8; ++i) { a[i] = (_Float16)p0[i]; a[8 + i] = (_Float16)p1[i]; }
  return a;
}

__device__ __forceinline__ v16h a_frag_h(const _Float16* p, int hi) {
  v16h a;
  const _Float16* p0 = p + 16 * hi;
  const _Float16* p1 = p0 + 8;
#pragma unroll
  for (int i = 0; i < 8; ++i) { a[i] = p0[i]; a[8 + i] = p1[i]; }
  return a;
}

// Single-tile pipelined GEMM (used where registers are tight, e.g. GRU).
template <int KB>
__device__ __forceinline__ v8f gemm_tile(const v16h (&A)[KB], const _Float16* Wt,
                                         int K, int n, int hi) {
  const _Float16* bp = Wt + (size_t)n * K + 16 * hi;
  v16h bnext = *(const v16h*)bp;
  v8f c = {};
#pragma unroll
  for (int kb = 0; kb < KB; ++kb) {
    v16h bc = bnext;
    if (kb + 1 < KB) bnext = *(const v16h*)(bp + (size_t)(kb + 1) * 32);
    c = wmma16(A[kb], bc, c);
  }
  return c;
}

// Dual-tile GEMM: two independent WMMA accumulation chains (cols n0 and n1)
// with interleaved B loads -> each wait covers two WMMAs and the next pair of
// fragment loads overlaps the current WMMA pair.
template <int KB>
__device__ __forceinline__ void gemm_tile2(const v16h (&A)[KB], const _Float16* Wt,
                                           int K, int n0, int n1, int hi,
                                           v8f& c0, v8f& c1) {
  const _Float16* bp0 = Wt + (size_t)n0 * K + 16 * hi;
  const _Float16* bp1 = Wt + (size_t)n1 * K + 16 * hi;
  v16h b0 = *(const v16h*)bp0;
  v16h b1 = *(const v16h*)bp1;
  c0 = {}; c1 = {};
#pragma unroll
  for (int kb = 0; kb < KB; ++kb) {
    v16h t0 = b0, t1 = b1;
    if (kb + 1 < KB) {
      b0 = *(const v16h*)(bp0 + (size_t)(kb + 1) * 32);
      b1 = *(const v16h*)(bp1 + (size_t)(kb + 1) * 32);
    }
    c0 = wmma16(A[kb], t0, c0);
    c1 = wmma16(A[kb], t1, c1);
  }
}

__device__ __forceinline__ float sigm(float x) { return 1.f / (1.f + __expf(-x)); }

// ---------------- setup kernels ----------------

__global__ void k_convert(_Float16* dst, const float* src, int K, int Nout, int transp) {
  int idx = blockIdx.x * blockDim.x + threadIdx.x;
  if (idx >= K * Nout) return;
  int n = idx / K, k = idx - n * K;
  dst[idx] = (_Float16)(transp ? src[(size_t)k * Nout + n] : src[idx]);
}

__global__ void k_cpl(const int* ei, const unsigned char* tie, float* cpl,
                      int* tieCount, int* tieList) {
  int e = blockIdx.x * blockDim.x + threadIdx.x;
  if (e >= EE) return;
  if (tie[e]) {
    atomicAdd(&cpl[ei[e]], 1.f);
    atomicAdd(&cpl[ei[EE + e]], 1.f);
    int pos = atomicAdd(tieCount, 1);
    tieList[pos] = e;
  }
}

__global__ void k_w(const int* ei, const unsigned char* tie, const float* ear,
                    const float* cpl, float* wden, float* wnorm) {
  int e = blockIdx.x * blockDim.x + threadIdx.x;
  if (e >= EE) return;
  int col = ei[EE + e];
  float w = 0.f;
  if (!tie[e] && cpl[col] > 0.f) {
    float X = ear[(size_t)e * 10 + 1];
    w = 1.f / sqrtf(X * X + 1e-6f);
    atomicAdd(&wden[col], w);
  }
  wnorm[e] = w;
}

__global__ void k_wnorm(const int* ei, const float* wden, float* wnorm) {
  int e = blockIdx.x * blockDim.x + threadIdx.x;
  if (e >= EE) return;
  float w = wnorm[e];
  if (w != 0.f) wnorm[e] = w / (wden[ei[EE + e]] + 1e-6f);
}

// ---------------- scatter kernels ----------------

// S[col] += w_norm * h[row]   (32 lanes per edge, 4 cols each)
__global__ void k_scatter_S(const int* ei, const float* wnorm, const float* hOut, float* S) {
  size_t t = (size_t)blockIdx.x * blockDim.x + threadIdx.x;
  int e = (int)(t >> 5);
  if (e >= EE) return;
  float w = wnorm[e];
  if (w == 0.f) return;
  int c0 = ((int)t & 31) * 4;
  const float* hp = hOut + (size_t)ei[e] * HD + c0;
  float* sp = S + (size_t)ei[EE + e] * HD + c0;
#pragma unroll
  for (int j = 0; j < 4; ++j) atomicAdd(&sp[j], w * hp[j]);
}

// m_edge[row] += e/deg[row]; m_edge[col] += e/deg[col]   on tie edges
__global__ void k_scatter_tie(const int* ei, const int* tieCount, const int* tieList,
                              const float* cpl, const float* eOut, float* medge) {
  size_t t = (size_t)blockIdx.x * blockDim.x + threadIdx.x;
  int et = (int)(t >> 5);
  if (et >= *tieCount) return;
  int e = tieList[et];
  int c0 = ((int)t & 31) * 4;
  int rd = ei[e], cd = ei[EE + e];
  float dr = fmaxf(cpl[rd], 1.f), dc = fmaxf(cpl[cd], 1.f);
  const float* ep = eOut + (size_t)e * HD + c0;
  float* mr = medge + (size_t)rd * HD + c0;
  float* mc = medge + (size_t)cd * HD + c0;
#pragma unroll
  for (int j = 0; j < 4; ++j) {
    float v = ep[j];
    atomicAdd(&mr[j], v / dr);
    atomicAdd(&mc[j], v / dc);
  }
}

// ---------------- WMMA GEMM kernels ----------------

// Tie-edge MLP: e += relu([e|h_row|h_col] @ Wer1 + b1) @ Wer2 + b2, tie edges only.
__global__ void k_edge_mlp(const int* ei, const int* tieCount, const int* tieList,
                           const float* hOut, float* eOut,
                           const _Float16* Wer1t, const float* ber1,
                           const _Float16* Wer2t, const float* ber2) {
  extern __shared__ _Float16 smem[];
  int lane = threadIdx.x & 31, wave = threadIdx.x >> 5;
  int tile = blockIdx.x * (blockDim.x >> 5) + wave;
  int Et = *tieCount;
  if (tile * 16 >= Et) return;          // no barriers used: early exit is safe
  int nl = lane & 15, hi = lane >> 4;
  _Float16* hid = smem + (size_t)wave * 16 * HSTRIDE;

  int r = tile * 16 + nl; if (r >= Et) r = Et - 1;
  int eIdx = tieList[r];
  const float* pe  = eOut + (size_t)eIdx * HD;
  const float* ph1 = hOut + (size_t)ei[eIdx] * HD;
  const float* ph2 = hOut + (size_t)ei[EE + eIdx] * HD;

  v16h A[12];
#pragma unroll
  for (int kb = 0; kb < 12; ++kb) {
    const float* src = (kb < 4) ? pe : ((kb < 8) ? ph1 : ph2);
    A[kb] = a_frag_f32(src + (kb & 3) * 32, hi);
  }

  // layer 1 (384 -> 256), relu, stage f16 in LDS; two tiles per pass
  for (int ntp = 0; ntp < 8; ++ntp) {
    int n0 = ntp * 32 + nl, n1 = n0 + 16;
    v8f c0, c1;
    gemm_tile2<12>(A, Wer1t, HD3, n0, n1, hi, c0, c1);
    float b0 = ber1[n0], b1 = ber1[n1];
#pragma unroll
    for (int vr = 0; vr < 8; ++vr) {
      float v0 = c0[vr] + b0; v0 = v0 > 0.f ? v0 : 0.f;
      float v1 = c1[vr] + b1; v1 = v1 > 0.f ? v1 : 0.f;
      _Float16* hrow = hid + (size_t)(vr + 8 * hi) * HSTRIDE;
      hrow[n0] = (_Float16)v0;
      hrow[n1] = (_Float16)v1;
    }
  }
  asm volatile("s_wait_dscnt 0" ::: "memory");   // LDS store -> load ordering within wave

  v16h A2[8];
#pragma unroll
  for (int kb = 0; kb < 8; ++kb) A2[kb] = a_frag_h(hid + (size_t)nl * HSTRIDE + kb * 32, hi);

  float* prow[8]; bool rv[8];
#pragma unroll
  for (int vr = 0; vr < 8; ++vr) {
    int rr = tile * 16 + vr + 8 * hi;
    rv[vr] = rr < Et;
    prow[vr] = eOut + (size_t)tieList[rv[vr] ? rr : 0] * HD;
  }

  // layer 2 (256 -> 128), accumulate into e_work; two tiles per pass
  for (int ntp = 0; ntp < 4; ++ntp) {
    int n0 = ntp * 32 + nl, n1 = n0 + 16;
    v8f c0, c1;
    gemm_tile2<8>(A2, Wer2t, HD2, n0, n1, hi, c0, c1);
    float b0 = ber2[n0], b1 = ber2[n1];
#pragma unroll
    for (int vr = 0; vr < 8; ++vr) {
      if (rv[vr]) {
        prow[vr][n0] += c0[vr] + b0;
        prow[vr][n1] += c1[vr] + b1;
      }
    }
  }
}

// m_node = S @ Wnn + wsum * b_nn
__global__ void k_mnode(const float* S, const float* wden, const _Float16* Wnnt,
                        const float* bnn, float* mnode) {
  int lane = threadIdx.x & 31, wave = threadIdx.x >> 5;
  int tile = blockIdx.x * (blockDim.x >> 5) + wave;
  if (tile * 16 >= NN) return;
  int nl = lane & 15, hi = lane >> 4;
  int row = tile * 16 + nl; if (row >= NN) row = NN - 1;
  const float* sp = S + (size_t)row * HD;
  v16h A[4];
#pragma unroll
  for (int kb = 0; kb < 4; ++kb) A[kb] = a_frag_f32(sp + kb * 32, hi);

  float ws8[8]; bool rv[8];
#pragma unroll
  for (int vr = 0; vr < 8; ++vr) {
    int rr = tile * 16 + vr + 8 * hi;
    rv[vr] = rr < NN;
    float wd = wden[rv[vr] ? rr : (NN - 1)];
    ws8[vr] = wd / (wd + 1e-6f);
  }
  for (int ntp = 0; ntp < 4; ++ntp) {
    int n0 = ntp * 32 + nl, n1 = n0 + 16;
    v8f c0, c1;
    gemm_tile2<4>(A, Wnnt, HD, n0, n1, hi, c0, c1);
    float b0 = bnn[n0], b1 = bnn[n1];
#pragma unroll
    for (int vr = 0; vr < 8; ++vr) {
      if (rv[vr]) {
        float* mrow = mnode + (size_t)(tile * 16 + vr + 8 * hi) * HD;
        mrow[n0] = c0[vr] + ws8[vr] * b0;
        mrow[n1] = c1[vr] + ws8[vr] * b1;
      }
    }
  }
}

// gate = sigmoid(relu([m_edge|m_node|h] @ Wg1 + b1) @ Wg2 + b2)
__global__ void k_gate(const float* medge, const float* mnode, const float* hOut,
                       const _Float16* Wg1t, const float* bg1,
                       const float* Wg2, const float* bg2, float* gate) {
  int lane = threadIdx.x & 31, wave = threadIdx.x >> 5;
  int tile = blockIdx.x * (blockDim.x >> 5) + wave;
  if (tile * 16 >= NN) return;
  int nl = lane & 15, hi = lane >> 4;
  int row = tile * 16 + nl; if (row >= NN) row = NN - 1;
  const float* p0 = medge + (size_t)row * HD;
  const float* p1 = mnode + (size_t)row * HD;
  const float* p2 = hOut + (size_t)row * HD;
  v16h A[12];
#pragma unroll
  for (int kb = 0; kb < 12; ++kb) {
    const float* src = (kb < 4) ? p0 : ((kb < 8) ? p1 : p2);
    A[kb] = a_frag_f32(src + (kb & 3) * 32, hi);
  }
  float acc[8];
#pragma unroll
  for (int vr = 0; vr < 8; ++vr) acc[vr] = 0.f;
  for (int ntp = 0; ntp < 4; ++ntp) {
    int n0 = ntp * 32 + nl, n1 = n0 + 16;
    v8f c0, c1;
    gemm_tile2<12>(A, Wg1t, HD3, n0, n1, hi, c0, c1);
    float b0 = bg1[n0], w20 = Wg2[n0];
    float b1 = bg1[n1], w21 = Wg2[n1];
#pragma unroll
    for (int vr = 0; vr < 8; ++vr) {
      float g0 = c0[vr] + b0; g0 = g0 > 0.f ? g0 : 0.f;
      float g1 = c1[vr] + b1; g1 = g1 > 0.f ? g1 : 0.f;
      acc[vr] += g0 * w20 + g1 * w21;
    }
  }
  // reduce across the 16 lanes of each half (rows vr+8*hi)
#pragma unroll
  for (int vr = 0; vr < 8; ++vr) {
#pragma unroll
    for (int m = 1; m < 16; m <<= 1) acc[vr] += __shfl_xor(acc[vr], m, 32);
  }
  if (nl == 0) {
    float b2 = bg2[0];
#pragma unroll
    for (int vr = 0; vr < 8; ++vr) {
      int rr = tile * 16 + vr + 8 * hi;
      if (rr < NN) gate[rr] = sigm(acc[vr] + b2);
    }
  }
}

// GRU update on coupling nodes, m = m_edge + gate*m_node built on the fly
__global__ void k_gru(const float* medge, const float* mnode, const float* gate,
                      const float* cpl, const float* bih, const float* bhh,
                      const _Float16* Wih, const _Float16* Whh, float* hOut) {
  int lane = threadIdx.x & 31, wave = threadIdx.x >> 5;
  int tile = blockIdx.x * (blockDim.x >> 5) + wave;
  if (tile * 16 >= NN) return;
  int nl = lane & 15, hi = lane >> 4;
  int row = tile * 16 + nl; if (row >= NN) row = NN - 1;
  float g = gate[row];
  const float* pme = medge + (size_t)row * HD;
  const float* pmn = mnode + (size_t)row * HD;
  const float* ph  = hOut + (size_t)row * HD;

  v16h M[4], Hf[4];
#pragma unroll
  for (int kb = 0; kb < 4; ++kb) {
    v16h a;
    int base = kb * 32 + 16 * hi;
#pragma unroll
    for (int i = 0; i < 8; ++i) {
      int k0 = base + i, k1 = base + 8 + i;
      a[i]     = (_Float16)(pme[k0] + g * pmn[k0]);
      a[8 + i] = (_Float16)(pme[k1] + g * pmn[k1]);
    }
    M[kb] = a;
    Hf[kb] = a_frag_f32(ph + kb * 32, hi);
  }

  bool cplv[8];
#pragma unroll
  for (int vr = 0; vr < 8; ++vr) {
    int rr = tile * 16 + vr + 8 * hi;
    cplv[vr] = (rr < NN) && (cpl[rr < NN ? rr : (NN - 1)] > 0.f);
  }

  v8f Rt[8], Zt[8];
  for (int nt = 0; nt < 8; ++nt) {           // r chunk (cols 0..127)
    int col = nt * 16 + nl;
    v8f ci = gemm_tile<4>(M, Wih, HD, col, hi);
    v8f ch = gemm_tile<4>(Hf, Whh, HD, col, hi);
    float bi = bih[col], bh = bhh[col];
    v8f r;
#pragma unroll
    for (int vr = 0; vr < 8; ++vr) r[vr] = sigm(ci[vr] + ch[vr] + bi + bh);
    Rt[nt] = r;
  }
  for (int nt = 0; nt < 8; ++nt) {           // z chunk (cols 128..255)
    int col = 128 + nt * 16 + nl;
    v8f ci = gemm_tile<4>(M, Wih, HD, col, hi);
    v8f ch = gemm_tile<4>(Hf, Whh, HD, col, hi);
    float bi = bih[col], bh = bhh[col];
    v8f z;
#pragma unroll
    for (int vr = 0; vr < 8; ++vr) z[vr] = sigm(ci[vr] + ch[vr] + bi + bh);
    Zt[nt] = z;
  }
  for (int nt = 0; nt < 8; ++nt) {           // n chunk (cols 256..383) + update
    int col = 256 + nt * 16 + nl;
    v8f ci = gemm_tile<4>(M, Wih, HD, col, hi);
    v8f ch = gemm_tile<4>(Hf, Whh, HD, col, hi);
    float bi = bih[col], bh = bhh[col];
    int cl = nt * 16 + nl;
#pragma unroll
    for (int vr = 0; vr < 8; ++vr) {
      if (cplv[vr]) {
        int rr = tile * 16 + vr + 8 * hi;
        float nv = tanhf(ci[vr] + bi + Rt[nt][vr] * (ch[vr] + bh));
        float z = Zt[nt][vr];
        size_t off = (size_t)rr * HD + cl;
        float ho = hOut[off];
        hOut[off] = (1.f - z) * nv + z * ho;
      }
    }
  }
}

// ---------------- host ----------------

static inline size_t al256(size_t x) { return (x + 255) & ~(size_t)255; }

extern "C" void kernel_launch(void* const* d_in, const int* in_sizes, int n_in,
                              void* d_out, int out_size, void* d_ws, size_t ws_size,
                              hipStream_t stream) {
  (void)in_sizes; (void)n_in; (void)out_size; (void)ws_size;
  const float* h    = (const float*)d_in[0];
  const float* e    = (const float*)d_in[1];
  const float* ear  = (const float*)d_in[2];
  const float* Wer1 = (const float*)d_in[3];
  const float* ber1 = (const float*)d_in[4];
  const float* Wer2 = (const float*)d_in[5];
  const float* ber2 = (const float*)d_in[6];
  const float* Wih  = (const float*)d_in[7];
  const float* Whh  = (const float*)d_in[8];
  const float* bih  = (const float*)d_in[9];
  const float* bhh  = (const float*)d_in[10];
  const float* Wnn  = (const float*)d_in[11];
  const float* bnn  = (const float*)d_in[12];
  const float* Wg1  = (const float*)d_in[13];
  const float* bg1  = (const float*)d_in[14];
  const float* Wg2  = (const float*)d_in[15];
  const float* bg2  = (const float*)d_in[16];
  const int* ei     = (const int*)d_in[17];
  const unsigned char* tie = (const unsigned char*)d_in[18];

  float* hOut = (float*)d_out;
  float* eOut = hOut + (size_t)NN * HD;

  // workspace carve
  char* w = (char*)d_ws; size_t off = 0;
  _Float16* Wer1t = (_Float16*)(w + off); off = al256(off + (size_t)HD2 * HD3 * 2);
  _Float16* Wer2t = (_Float16*)(w + off); off = al256(off + (size_t)HD * HD2 * 2);
  _Float16* Wnnt  = (_Float16*)(w + off); off = al256(off + (size_t)HD * HD * 2);
  _Float16* Wg1t  = (_Float16*)(w + off); off = al256(off + (size_t)HD * HD3 * 2);
  _Float16* Wihh  = (_Float16*)(w + off); off = al256(off + (size_t)HD3 * HD * 2);
  _Float16* Whhh  = (_Float16*)(w + off); off = al256(off + (size_t)HD3 * HD * 2);
  float* cpl   = (float*)(w + off); off = al256(off + (size_t)NN * 4);
  float* wden  = (float*)(w + off); off = al256(off + (size_t)NN * 4);
  float* gatep = (float*)(w + off); off = al256(off + (size_t)NN * 4);
  float* wnorm = (float*)(w + off); off = al256(off + (size_t)EE * 4);
  int* tieCnt  = (int*)(w + off);   off = al256(off + 256);
  int* tieList = (int*)(w + off);   off = al256(off + (size_t)EE * 4);
  float* S     = (float*)(w + off); off = al256(off + (size_t)NN * HD * 4);
  float* medge = (float*)(w + off); off = al256(off + (size_t)NN * HD * 4);
  float* mnode = (float*)(w + off); off = al256(off + (size_t)NN * HD * 4);

  // working copies of h, e in d_out (updated in place; final state == output)
  hipMemcpyAsync(hOut, h, (size_t)NN * HD * 4, hipMemcpyDeviceToDevice, stream);
  hipMemcpyAsync(eOut, e, (size_t)EE * HD * 4, hipMemcpyDeviceToDevice, stream);

  // weight conversion to f16 N-major
  auto cvblk = [](int n) { return (n + 255) / 256; };
  k_convert<<<cvblk(HD3 * HD2), 256, 0, stream>>>(Wer1t, Wer1, HD3, HD2, 1);
  k_convert<<<cvblk(HD2 * HD), 256, 0, stream>>>(Wer2t, Wer2, HD2, HD, 1);
  k_convert<<<cvblk(HD * HD), 256, 0, stream>>>(Wnnt, Wnn, HD, HD, 1);
  k_convert<<<cvblk(HD3 * HD), 256, 0, stream>>>(Wg1t, Wg1, HD3, HD, 1);
  k_convert<<<cvblk(HD3 * HD), 256, 0, stream>>>(Wihh, Wih, HD, HD3, 0);
  k_convert<<<cvblk(HD3 * HD), 256, 0, stream>>>(Whhh, Whh, HD, HD3, 0);

  // iteration-invariant graph stats
  hipMemsetAsync(cpl, 0, (size_t)NN * 4, stream);
  hipMemsetAsync(wden, 0, (size_t)NN * 4, stream);
  hipMemsetAsync(tieCnt, 0, 256, stream);
  int eblk = (EE + 255) / 256;
  k_cpl<<<eblk, 256, 0, stream>>>(ei, tie, cpl, tieCnt, tieList);
  k_w<<<eblk, 256, 0, stream>>>(ei, tie, ear, cpl, wden, wnorm);
  k_wnorm<<<eblk, 256, 0, stream>>>(ei, wden, wnorm);

  const int nodeTiles = (NN + 15) / 16;
  const int nodeBlocks = (nodeTiles + 7) / 8;
  const int mlpBlocks = ((EE / 16) + 3) / 4;          // worst case; waves exit on Et
  const size_t mlpLds = (size_t)4 * 16 * HSTRIDE * 2; // 33792 B
  const int scatBlocks = (int)(((size_t)EE * 32 + 255) / 256);

  for (int it = 0; it < NITER; ++it) {
    hipMemsetAsync(S, 0, (size_t)NN * HD * 4, stream);
    hipMemsetAsync(medge, 0, (size_t)NN * HD * 4, stream);

    k_edge_mlp<<<mlpBlocks, 128, mlpLds, stream>>>(ei, tieCnt, tieList, hOut, eOut,
                                                   Wer1t, ber1, Wer2t, ber2);
    k_scatter_S<<<scatBlocks, 256, 0, stream>>>(ei, wnorm, hOut, S);
    k_scatter_tie<<<scatBlocks, 256, 0, stream>>>(ei, tieCnt, tieList, cpl, eOut, medge);
    k_mnode<<<nodeBlocks, 256, 0, stream>>>(S, wden, Wnnt, bnn, mnode);
    k_gate<<<nodeBlocks, 256, 0, stream>>>(medge, mnode, hOut, Wg1t, bg1, Wg2, bg2, gatep);
    k_gru<<<nodeBlocks, 256, 0, stream>>>(medge, mnode, gatep, cpl, bih, bhh,
                                          Wihh, Whhh, hOut);
  }
}
